// TailNet_14714557956196
// MI455X (gfx1250) — matrix operand order repaired and checked by reading, not compiled, not run on hardware
//
#include <hip/hip_runtime.h>

typedef __attribute__((ext_vector_type(16))) __bf16 v16bf;
typedef __attribute__((ext_vector_type(8)))  __bf16 v8bf;
typedef __attribute__((ext_vector_type(8)))  float  v8f;

#define B_    8
#define CIN   64
#define H_    256
#define W_    256
#define COUT  3
#define KSZ   7
#define PAD   3

#define ROWS     2                            // output rows per block (per wave: 2 accumulators)
#define NIY      (KSZ + ROWS - 1)             // 8 staged input rows
#define CICHUNK  32
#define NCHUNK   (CIN / CICHUNK)              // 2 ci-chunks of 32 (one WMMA K each)
#define IXS      264                          // staged row width: 262 used (-3..258), pad to 264
#define ROWSTRIDE (IXS * CICHUNK)             // elements per staged input row = 8448
#define XS_ELEMS (NIY * ROWSTRIDE)            // 8*264*32 = 67584 bf16
#define FS_ELEMS (KSZ * KSZ * COUT * CICHUNK) // 49*3*32  = 4704 bf16
#define LDS_ELEMS (XS_ELEMS + FS_ELEMS)       // 72288 bf16 -> 144576 bytes

// round-to-nearest-even f32 -> bf16 (bit trick)
__device__ __forceinline__ unsigned short f2bf(float f) {
    unsigned int u = __builtin_bit_cast(unsigned int, f);
    u += 0x7FFFu + ((u >> 16) & 1u);
    return (unsigned short)(u >> 16);
}

__global__ void __launch_bounds__(512)
conv7x7_circ_wmma(const float* __restrict__ x, const float* __restrict__ flt,
                  float* __restrict__ out) {
    extern __shared__ unsigned short smem[];
    unsigned short* xs = smem;              // [iy(8)][ixs(264)][ci(32)] bf16
    unsigned short* fs = smem + XS_ELEMS;   // [p=kh*7+kw(49)][co(3)][ci(32)] bf16

    const int tid  = threadIdx.x;
    const int lane = tid & 31;
    const int wave = tid >> 5;              // 0..15 -> 16-pixel tile along W
    const int bid  = blockIdx.x;
    const int b    = bid >> 7;              // / (H_/ROWS)
    const int oy0  = (bid & (H_ / ROWS - 1)) * ROWS;

    const int ox_base = wave * 16;
    const int m    = lane & 15;             // A-matrix row (output pixel within tile)
    const int hi   = lane >> 4;             // lane half selects K-groups per ISA layout
    const int koff = hi * 8;                // A: lanes<16 hold K{0..7,16..23}; >=16 hold K{8..15,24..31}

    const int nco        = lane & 15;       // B/D column = cout slot
    const int co_clamped = nco < COUT ? nco : (COUT - 1); // garbage cols read valid memory

    v8f acc0 = {};                          // output row oy0
    v8f acc1 = {};                          // output row oy0+1

    for (int cc = 0; cc < NCHUNK; ++cc) {
        const int ci0 = cc * CICHUNK;
        __syncthreads();  // previous chunk's compute done before LDS overwrite

        // ---- stage X: rows oy0-3..oy0+4 (circular), cols ix = ixs-3 (circular), bf16, ci-fastest
        for (int idx = tid; idx < XS_ELEMS; idx += 512) {
            int iy  = idx / ROWSTRIDE;
            int rem = idx - iy * ROWSTRIDE;
            int cil = rem / IXS;
            int ixs = rem - cil * IXS;
            int iy_g = (oy0 + iy - PAD + H_) & (H_ - 1);
            int ix_g = (ixs - PAD + W_) & (W_ - 1);
            float v = x[(((size_t)b * CIN + (ci0 + cil)) * H_ + iy_g) * W_ + ix_g];
            xs[(iy * IXS + ixs) * CICHUNK + cil] = f2bf(v);
        }
        // ---- stage filters for this batch & ci-chunk: fs[p][co][cil]
        for (int idx = tid; idx < FS_ELEMS; idx += 512) {
            int p   = idx / (COUT * CICHUNK);
            int rem = idx - p * (COUT * CICHUNK);
            int co  = rem / CICHUNK;
            int cil = rem - co * CICHUNK;
            int kh  = p / KSZ;
            int kw  = p - kh * KSZ;
            float v = flt[((((size_t)b * COUT + co) * CIN + (ci0 + cil)) * KSZ + kh) * KSZ + kw];
            fs[idx] = f2bf(v);
        }
        __syncthreads();

        // ---- 49 x 2 WMMAs: K=32 cin per (kh,kw); B fragment shared by both output rows
        #pragma unroll 7
        for (int p = 0; p < KSZ * KSZ; ++p) {
            const int kh  = p / KSZ;
            const int kw  = p - kh * KSZ;
            const int ixs = ox_base + m + kw;   // = (ox + kw - 3) + PAD, 0..261

            // B fragment: W(co, ci0..ci0+31, kh, kw); lanes<16 hold K{0..7,16..23}, >=16 hold K{8..15,24..31}
            const unsigned short* fb = &fs[(p * COUT + co_clamped) * CICHUNK + hi * 8];
            v8bf b_lo = *reinterpret_cast<const v8bf*>(fb);
            v8bf b_hi = *reinterpret_cast<const v8bf*>(fb + 16);
            v16bf bfr = __builtin_shufflevector(b_lo, b_hi,
                0,1,2,3,4,5,6,7,8,9,10,11,12,13,14,15);

            // A fragment, output row oy0: staged input row index = kh
            const unsigned short* xb0 = &xs[(kh * IXS + ixs) * CICHUNK + koff];
            v8bf a0_lo = *reinterpret_cast<const v8bf*>(xb0);
            v8bf a0_hi = *reinterpret_cast<const v8bf*>(xb0 + 16);
            v16bf a0 = __builtin_shufflevector(a0_lo, a0_hi,
                0,1,2,3,4,5,6,7,8,9,10,11,12,13,14,15);
            acc0 = __builtin_amdgcn_wmma_f32_16x16x32_bf16(
                false, a0, false, bfr, (short)0, acc0, false, false);

            // A fragment, output row oy0+1: staged input row index = kh+1 (addr + ROWSTRIDE)
            const unsigned short* xb1 = xb0 + ROWSTRIDE;
            v8bf a1_lo = *reinterpret_cast<const v8bf*>(xb1);
            v8bf a1_hi = *reinterpret_cast<const v8bf*>(xb1 + 16);
            v16bf a1 = __builtin_shufflevector(a1_lo, a1_hi,
                0,1,2,3,4,5,6,7,8,9,10,11,12,13,14,15);
            acc1 = __builtin_amdgcn_wmma_f32_16x16x32_bf16(
                false, a1, false, bfr, (short)0, acc1, false, false);
        }
    }

    // ---- store: lane holds D column n = lane&15 (cout), rows m0..m0+7 in acc[0..7]
    if (nco < COUT) {
        const int m0 = hi * 8;
        float* op0 = out + ((((size_t)b * COUT + nco) * H_ + oy0) * W_ + ox_base + m0);
        *reinterpret_cast<float4*>(op0)     = make_float4(acc0[0], acc0[1], acc0[2], acc0[3]);
        *reinterpret_cast<float4*>(op0 + 4) = make_float4(acc0[4], acc0[5], acc0[6], acc0[7]);
        float* op1 = op0 + W_;  // row oy0+1
        *reinterpret_cast<float4*>(op1)     = make_float4(acc1[0], acc1[1], acc1[2], acc1[3]);
        *reinterpret_cast<float4*>(op1 + 4) = make_float4(acc1[4], acc1[5], acc1[6], acc1[7]);
    }
}

extern "C" void kernel_launch(void* const* d_in, const int* in_sizes, int n_in,
                              void* d_out, int out_size, void* d_ws, size_t ws_size,
                              hipStream_t stream) {
    (void)in_sizes; (void)n_in; (void)out_size; (void)d_ws; (void)ws_size;
    const float* x = (const float*)d_in[0];   // [8,64,256,256] f32
    const float* f = (const float*)d_in[1];   // [8,3,64,7,7]   f32
    float* out = (float*)d_out;               // [8,3,256,256]  f32

    dim3 grid(B_ * H_ / ROWS);  // 1024 workgroups: one per (batch, pair of output rows)
    dim3 block(512);            // 16 wave32s: one 16-pixel WMMA tile x 2 rows each
    size_t lds = (size_t)LDS_ELEMS * sizeof(unsigned short);  // 144,576 B
    conv7x7_circ_wmma<<<grid, block, lds, stream>>>(x, f, out);
}
